// CTBTagger_54915451846909
// MI455X (gfx1250) — matrix operand order repaired and checked by reading, not compiled, not run on hardware
//
#include <hip/hip_runtime.h>
#include <hip/hip_bf16.h>

#define NW 8192      // words
#define LCC 16       // chars per word
#define CDIM 100     // char emb dim
#define CDP 128      // padded char emb dim
#define WDIM 300     // word emb dim
#define HCH 256      // per-direction hidden
#define G4 1024      // 4*H
#define WXD 812      // WD + HD
#define WXP 832      // padded
#define T1N 48
#define T2N 32

typedef __bf16 bf16_t;
typedef bf16_t v16bf __attribute__((ext_vector_type(16)));
typedef float  v8f   __attribute__((ext_vector_type(8)));

// POD builtin vectors (HIP's uint4/float4 are classes whose copy ctors cannot
// bind address_space(1) references -- ext_vector types have no ctors).
typedef unsigned int u32x4 __attribute__((ext_vector_type(4)));
typedef float        f32x4 __attribute__((ext_vector_type(4)));

union FragAB {
    v16bf v;
    u32x4 q[2];
    unsigned short us[16];
};

__device__ __forceinline__ unsigned short f2bf(float f) {
    union { float f; unsigned int u; } cv; cv.f = f;
    unsigned int u = cv.u;
    u += 0x7fffu + ((u >> 16) & 1u);     // round-to-nearest-even
    return (unsigned short)(u >> 16);
}

__device__ __forceinline__ float sigm(float x) { return 1.f / (1.f + __expf(-x)); }

// Make a pointer opaque to the optimizer: prevents LICM from hoisting the
// (loop-invariant) weight-fragment loads out of the recurrence loop, which
// in round 1 caused them to be spilled to scratch and reloaded per WMMA.
template <typename T>
__device__ __forceinline__ T* opaque_ptr(T* p) {
    asm volatile("" : "+v"(p));
    return p;
}

// Loads through an explicit global address-space pointer: the opaque asm
// round-trip loses address-space inference and round-2 codegen emitted
// flat_load_b128 (ties up DScnt + LDS path). Force global_load_b128.
typedef const u32x4 __attribute__((address_space(1))) g_u32x4;
typedef const f32x4 __attribute__((address_space(1))) g_f32x4;

__device__ __forceinline__ u32x4 gload_u4(const void* p) {
    return *(g_u32x4*)(unsigned long long)p;
}
__device__ __forceinline__ f32x4 gload_f4(const void* p) {
    return *(g_f32x4*)(unsigned long long)p;
}

// ---------------------------------------------------------------------------
// f32 -> bf16 with column padding
// ---------------------------------------------------------------------------
__global__ void conv_pad_bf16(const float* __restrict__ src, unsigned short* __restrict__ dst,
                              int rows, int scols, int dcols) {
    int idx = blockIdx.x * blockDim.x + threadIdx.x;
    if (idx >= rows * dcols) return;
    int r = idx / dcols, c = idx - r * dcols;
    float v = (c < scols) ? src[r * scols + c] : 0.f;
    dst[idx] = f2bf(v);
}

// ---------------------------------------------------------------------------
// Char-level BiLSTM: one block = 16 words x full 16-step scan, one direction.
// 8 waves; wave w owns h columns [w*32, w*32+32) => 8 gate N-tiles of 16.
// ---------------------------------------------------------------------------
__global__ __launch_bounds__(256)
void char_bilstm_kernel(const int* __restrict__ cseq,        // [NW, LCC]
                        const int* __restrict__ clens,       // [NW]
                        const float* __restrict__ cemb,      // [CV, CDIM]
                        const unsigned short* __restrict__ WihF,  // [G4, CDP]
                        const unsigned short* __restrict__ WhhF,  // [G4, HCH]
                        const unsigned short* __restrict__ WihB,
                        const unsigned short* __restrict__ WhhB,
                        const float* __restrict__ bihF, const float* __restrict__ bhhF,
                        const float* __restrict__ bihB, const float* __restrict__ bhhB,
                        unsigned short* __restrict__ char_feat)   // [NW, 512] bf16
{
    int dir  = blockIdx.x & 1;
    int tile = blockIdx.x >> 1;
    int w0   = tile * 16;

    const unsigned short* Wih = dir ? WihB : WihF;
    const unsigned short* Whh = dir ? WhhB : WhhF;
    const float* bih = dir ? bihB : bihF;
    const float* bhh = dir ? bhhB : bhhF;

    __shared__ __align__(16) unsigned short xtile[16][CDP];   // [word][k]  bf16
    __shared__ __align__(16) unsigned short hbuf[16][HCH];    // [word][col] bf16
    __shared__ int ids_sh[16];

    int tid  = threadIdx.x;
    int lane = tid & 31;
    int wave = tid >> 5;
    int hl   = lane >> 4;        // lane half (0/1)
    int nb   = lane & 15;

    // zero initial h
    for (int idx = tid; idx < 16 * HCH; idx += 256) ((unsigned short*)hbuf)[idx] = 0;

    // lengths for this lane's D rows (m = e + 8*hl)
    int lens[8];
#pragma unroll
    for (int e = 0; e < 8; ++e) lens[e] = clens[w0 + 8 * hl + e];

    // bias per (gate, subtile): column n is the same for all 8 D elements of a lane
    float bsum[4][2];
#pragma unroll
    for (int g = 0; g < 4; ++g)
#pragma unroll
        for (int j = 0; j < 2; ++j) {
            int n = g * 256 + wave * 32 + j * 16 + nb;
            bsum[g][j] = bih[n] + bhh[n];
        }

    v8f cacc[2];
#pragma unroll
    for (int j = 0; j < 2; ++j)
#pragma unroll
        for (int e = 0; e < 8; ++e) cacc[j][e] = 0.f;

    for (int step = 0; step < LCC; ++step) {
        int tc = dir ? (LCC - 1 - step) : step;   // reversed scan for backward dir

        if (tid < 16) ids_sh[tid] = cseq[(w0 + tid) * LCC + tc];
        __syncthreads();

        // stage x[t] tile (gather char embeddings, pad 100->128)
        for (int idx = tid; idx < 16 * CDP; idx += 256) {
            int m = idx >> 7, c = idx & 127;
            float v = (c < CDIM) ? cemb[ids_sh[m] * CDIM + c] : 0.f;
            xtile[m][c] = f2bf(v);
        }
        __syncthreads();

        // re-opaque the weight bases every step so fragment loads stay inside
        // the loop (stream from L2) instead of being hoisted + spilled.
        const unsigned short* WihS = opaque_ptr(Wih);
        const unsigned short* WhhS = opaque_ptr(Whh);

        // accumulators init with biases
        v8f acc[4][2];
#pragma unroll
        for (int g = 0; g < 4; ++g)
#pragma unroll
            for (int j = 0; j < 2; ++j)
#pragma unroll
                for (int e = 0; e < 8; ++e) acc[g][j][e] = bsum[g][j];

        // ---- x_t @ Wih.T  (K = 128) ----
#pragma unroll 1
        for (int kk = 0; kk < 4; ++kk) {
            FragAB a;
            int kb = kk * 32 + hl * 8;
            a.q[0] = *(const u32x4*)&xtile[nb][kb];
            a.q[1] = *(const u32x4*)&xtile[nb][kb + 16];
            int kb0 = kk * 32 + hl * 8;
#pragma unroll
            for (int g = 0; g < 4; ++g)
#pragma unroll
                for (int j = 0; j < 2; ++j) {
                    int nrow = g * 256 + wave * 32 + j * 16 + nb;
                    FragAB b;
                    const unsigned short* bp8 = WihS + nrow * CDP + kb0;
                    b.q[0] = gload_u4(bp8);
                    b.q[1] = gload_u4(bp8 + 16);
                    acc[g][j] = __builtin_amdgcn_wmma_f32_16x16x32_bf16(
                        false, a.v, false, b.v, (short)0, acc[g][j], false, false);
                }
        }

        // ---- h @ Whh.T  (K = 256) ----
#pragma unroll 1
        for (int kk = 0; kk < 8; ++kk) {
            FragAB a;
            int kb = kk * 32 + hl * 8;
            a.q[0] = *(const u32x4*)&hbuf[nb][kb];
            a.q[1] = *(const u32x4*)&hbuf[nb][kb + 16];
            int kb0 = kk * 32 + hl * 8;
#pragma unroll
            for (int g = 0; g < 4; ++g)
#pragma unroll
                for (int j = 0; j < 2; ++j) {
                    int nrow = g * 256 + wave * 32 + j * 16 + nb;
                    FragAB b;
                    const unsigned short* bp8 = WhhS + nrow * HCH + kb0;
                    b.q[0] = gload_u4(bp8);
                    b.q[1] = gload_u4(bp8 + 16);
                    acc[g][j] = __builtin_amdgcn_wmma_f32_16x16x32_bf16(
                        false, a.v, false, b.v, (short)0, acc[g][j], false, false);
                }
        }
        __syncthreads();   // all hbuf reads done before masked update

        // gate math + masked state update (i,f,g,o order)
#pragma unroll
        for (int j = 0; j < 2; ++j)
#pragma unroll
            for (int e = 0; e < 8; ++e) {
                int m = 8 * hl + e;
                bool msk = (tc < lens[e]);
                float iv = acc[0][j][e], fv = acc[1][j][e];
                float gv = acc[2][j][e], ov = acc[3][j][e];
                float cn = sigm(fv) * cacc[j][e] + sigm(iv) * tanhf(gv);
                float hn = sigm(ov) * tanhf(cn);
                if (msk) {
                    cacc[j][e] = cn;
                    hbuf[m][wave * 32 + j * 16 + nb] = f2bf(hn);
                }
            }
        // next iteration's barriers separate these writes from the next reads
    }
    __syncthreads();

    // final h (held at last valid step by mask) -> char_feat
    for (int idx = tid; idx < 16 * HCH; idx += 256) {
        int m = idx >> 8, c = idx & 255;
        char_feat[(w0 + m) * 512 + dir * 256 + c] = hbuf[m][c];
    }
}

// ---------------------------------------------------------------------------
// wx = [word_emb | char_feat | 0-pad] as bf16, [NW, 832]
// ---------------------------------------------------------------------------
__global__ void build_wx(const int* __restrict__ wseq, const float* __restrict__ wemb,
                         const unsigned short* __restrict__ char_feat,
                         unsigned short* __restrict__ wx) {
    int idx = blockIdx.x * blockDim.x + threadIdx.x;
    if (idx >= NW * WXP) return;
    int w = idx / WXP, c = idx - w * WXP;
    unsigned short v;
    if (c < WDIM)      v = f2bf(wemb[wseq[w] * WDIM + c]);
    else if (c < WXD)  v = char_feat[w * 512 + (c - WDIM)];
    else               v = 0;
    wx[idx] = v;
}

// ---------------------------------------------------------------------------
// xg = wx @ Wih.T + (bih+bhh), per direction.  Block: 16 M x 128 N, 8 waves.
// grid = 512 Mtiles * 8 Ngroups * 2 dirs
// ---------------------------------------------------------------------------
__global__ __launch_bounds__(256)
void word_input_gemm(const unsigned short* __restrict__ wx,
                     const unsigned short* __restrict__ WihF,
                     const unsigned short* __restrict__ WihB,
                     const float* __restrict__ bihF, const float* __restrict__ bhhF,
                     const float* __restrict__ bihB, const float* __restrict__ bhhB,
                     float* __restrict__ xgF, float* __restrict__ xgB) {
    int b   = blockIdx.x;
    int dir = b & 1;  b >>= 1;
    int ng  = b & 7;  b >>= 3;
    int m0  = b * 16;
    int lane = threadIdx.x & 31, wave = threadIdx.x >> 5;
    int hl = lane >> 4, nb = lane & 15;
    int n0 = ng * 128 + wave * 16;

    const unsigned short* Wih = dir ? WihB : WihF;
    const float* bih = dir ? bihB : bihF;
    const float* bhh = dir ? bhhB : bhhF;
    float* xg = dir ? xgB : xgF;

    v8f acc;
#pragma unroll
    for (int e = 0; e < 8; ++e) acc[e] = 0.f;

#pragma unroll 1
    for (int kk = 0; kk < WXP / 32; ++kk) {     // 26 k-steps
        FragAB a, bb;
        int row = m0 + nb;
        int kb  = kk * 32 + hl * 8;
        const unsigned short* ap = wx + row * WXP + kb;
        a.q[0] = *(const u32x4*)(ap);
        a.q[1] = *(const u32x4*)(ap + 16);
        int kb0 = kk * 32 + hl * 8;
        const unsigned short* bp8 = Wih + (n0 + nb) * WXP + kb0;
        bb.q[0] = *(const u32x4*)(bp8);
        bb.q[1] = *(const u32x4*)(bp8 + 16);
        acc = __builtin_amdgcn_wmma_f32_16x16x32_bf16(
            false, a.v, false, bb.v, (short)0, acc, false, false);
    }

    int n = n0 + nb;
    float bs = bih[n] + bhh[n];
#pragma unroll
    for (int e = 0; e < 8; ++e) {
        int m = m0 + e + 8 * hl;
        xg[m * G4 + n] = acc[e] + bs;
    }
}

// ---------------------------------------------------------------------------
// Word-level LSTM scan: batch=1, 8192 sequential steps. One block per dir.
// 1024 threads: thread r computes gate row r via 256-length dot (h in LDS).
// ---------------------------------------------------------------------------
__global__ __launch_bounds__(1024)
void word_lstm(const float* __restrict__ xgF, const float* __restrict__ xgB,
               const float* __restrict__ WhhF, const float* __restrict__ WhhB,
               float* __restrict__ out_hs) {                  // [NW, 512]
    int dir = blockIdx.x;
    const float* xg  = dir ? xgB  : xgF;
    const float* Whh = dir ? WhhB : WhhF;

    __shared__ __align__(16) float hvec[HCH];
    __shared__ float gvec[G4];

    int tid = threadIdx.x;
    float cst = 0.f;
    if (tid < HCH) hvec[tid] = 0.f;
    __syncthreads();

    for (int s = 0; s < NW; ++s) {
        int t = dir ? (NW - 1 - s) : s;
        float g = xg[t * G4 + tid];
        // opaque base: keep the 64 float4 weight loads inside the loop
        // (L1-resident) instead of hoisted into a ~256-VGPR spill.
        const float* wr = opaque_ptr(Whh + (size_t)tid * HCH);
#pragma unroll 8
        for (int k = 0; k < HCH / 4; ++k) {
            f32x4 w4 = gload_f4(wr + 4 * k);
            f32x4 h4 = *(const f32x4*)&hvec[4 * k];
            g = fmaf(w4.x, h4.x, g);
            g = fmaf(w4.y, h4.y, g);
            g = fmaf(w4.z, h4.z, g);
            g = fmaf(w4.w, h4.w, g);
        }
        gvec[tid] = g;
        __syncthreads();
        if (tid < HCH) {
            float iv = gvec[tid], fv = gvec[HCH + tid];
            float gg = gvec[2 * HCH + tid], ov = gvec[3 * HCH + tid];
            cst = sigm(fv) * cst + sigm(iv) * tanhf(gg);
            float h = sigm(ov) * tanhf(cst);
            hvec[tid] = h;
            out_hs[t * 512 + dir * 256 + tid] = h;
        }
        __syncthreads();
    }
}

// ---------------------------------------------------------------------------
// logits + log_softmax for both tagsets; one block per word.
// ---------------------------------------------------------------------------
__global__ __launch_bounds__(128)
void proj_logsoftmax(const float* __restrict__ out_hs,
                     const float* __restrict__ Wp,  const float* __restrict__ bp,
                     const float* __restrict__ Wp2, const float* __restrict__ bp2,
                     float* __restrict__ dout) {
    int w = blockIdx.x;
    __shared__ __align__(16) float rowv[512];
    __shared__ float logits[T1N + T2N];
    __shared__ float mx[2], ls[2];
    int tid = threadIdx.x;

    for (int i = tid; i < 512; i += 128) rowv[i] = out_hs[w * 512 + i];
    __syncthreads();

    if (tid < T1N + T2N) {
        const float* Wrow; float b;
        if (tid < T1N) { Wrow = Wp + tid * 512;          b = bp[tid]; }
        else           { Wrow = Wp2 + (tid - T1N) * 512; b = bp2[tid - T1N]; }
        float acc = b;
        for (int k = 0; k < 512; ++k) acc = fmaf(Wrow[k], rowv[k], acc);
        logits[tid] = acc;
    }
    __syncthreads();

    if (tid == 0) {
        float m = -3.4e38f;
        for (int i = 0; i < T1N; ++i) m = fmaxf(m, logits[i]);
        float s = 0.f;
        for (int i = 0; i < T1N; ++i) s += __expf(logits[i] - m);
        mx[0] = m; ls[0] = logf(s);
    }
    if (tid == 1) {
        float m = -3.4e38f;
        for (int i = T1N; i < T1N + T2N; ++i) m = fmaxf(m, logits[i]);
        float s = 0.f;
        for (int i = T1N; i < T1N + T2N; ++i) s += __expf(logits[i] - m);
        mx[1] = m; ls[1] = logf(s);
    }
    __syncthreads();

    if (tid < T1N)
        dout[w * T1N + tid] = logits[tid] - mx[0] - ls[0];
    else if (tid < T1N + T2N)
        dout[NW * T1N + w * T2N + (tid - T1N)] = logits[tid] - mx[1] - ls[1];
}

// ---------------------------------------------------------------------------
extern "C" void kernel_launch(void* const* d_in, const int* in_sizes, int n_in,
                              void* d_out, int out_size, void* d_ws, size_t ws_size,
                              hipStream_t stream) {
    (void)in_sizes; (void)n_in; (void)out_size; (void)ws_size;

    const int*   wseq  = (const int*)d_in[0];
    const int*   cseq  = (const int*)d_in[1];
    const int*   clens = (const int*)d_in[2];
    const float* cemb  = (const float*)d_in[3];
    const float* wemb  = (const float*)d_in[4];
    const float* cfWih = (const float*)d_in[5];
    const float* cfWhh = (const float*)d_in[6];
    const float* cfbih = (const float*)d_in[7];
    const float* cfbhh = (const float*)d_in[8];
    const float* cbWih = (const float*)d_in[9];
    const float* cbWhh = (const float*)d_in[10];
    const float* cbbih = (const float*)d_in[11];
    const float* cbbhh = (const float*)d_in[12];
    const float* wfWih = (const float*)d_in[13];
    const float* wfWhh = (const float*)d_in[14];
    const float* wfbih = (const float*)d_in[15];
    const float* wfbhh = (const float*)d_in[16];
    const float* wbWih = (const float*)d_in[17];
    const float* wbWhh = (const float*)d_in[18];
    const float* wbbih = (const float*)d_in[19];
    const float* wbbhh = (const float*)d_in[20];
    const float* Wp    = (const float*)d_in[21];
    const float* bp    = (const float*)d_in[22];
    const float* Wp2   = (const float*)d_in[23];
    const float* bp2   = (const float*)d_in[24];

    // workspace carve (256B aligned)
    char* ws = (char*)d_ws;
    auto carve = [&](size_t bytes) -> void* {
        void* p = (void*)ws;
        ws += (bytes + 255) & ~(size_t)255;
        return p;
    };
    unsigned short* cWihF_bf = (unsigned short*)carve((size_t)G4 * CDP * 2);
    unsigned short* cWihB_bf = (unsigned short*)carve((size_t)G4 * CDP * 2);
    unsigned short* cWhhF_bf = (unsigned short*)carve((size_t)G4 * HCH * 2);
    unsigned short* cWhhB_bf = (unsigned short*)carve((size_t)G4 * HCH * 2);
    unsigned short* wWihF_bf = (unsigned short*)carve((size_t)G4 * WXP * 2);
    unsigned short* wWihB_bf = (unsigned short*)carve((size_t)G4 * WXP * 2);
    unsigned short* char_feat = (unsigned short*)carve((size_t)NW * 512 * 2);
    unsigned short* wx        = (unsigned short*)carve((size_t)NW * WXP * 2);
    float* xgF    = (float*)carve((size_t)NW * G4 * 4);
    float* xgB    = (float*)carve((size_t)NW * G4 * 4);
    float* out_hs = (float*)carve((size_t)NW * 512 * 4);

    // weight conversion
    conv_pad_bf16<<<(G4 * CDP + 255) / 256, 256, 0, stream>>>(cfWih, cWihF_bf, G4, CDIM, CDP);
    conv_pad_bf16<<<(G4 * CDP + 255) / 256, 256, 0, stream>>>(cbWih, cWihB_bf, G4, CDIM, CDP);
    conv_pad_bf16<<<(G4 * HCH + 255) / 256, 256, 0, stream>>>(cfWhh, cWhhF_bf, G4, HCH, HCH);
    conv_pad_bf16<<<(G4 * HCH + 255) / 256, 256, 0, stream>>>(cbWhh, cWhhB_bf, G4, HCH, HCH);
    conv_pad_bf16<<<(G4 * WXP + 255) / 256, 256, 0, stream>>>(wfWih, wWihF_bf, G4, WXD, WXP);
    conv_pad_bf16<<<(G4 * WXP + 255) / 256, 256, 0, stream>>>(wbWih, wWihB_bf, G4, WXD, WXP);

    // char-level BiLSTM (WMMA)
    char_bilstm_kernel<<<(NW / 16) * 2, 256, 0, stream>>>(
        cseq, clens, cemb,
        cWihF_bf, cWhhF_bf, cWihB_bf, cWhhB_bf,
        cfbih, cfbhh, cbbih, cbbhh,
        char_feat);

    // assemble wx and run word-level input GEMM (WMMA)
    build_wx<<<(NW * WXP + 255) / 256, 256, 0, stream>>>(wseq, wemb, char_feat, wx);
    word_input_gemm<<<(NW / 16) * 8 * 2, 256, 0, stream>>>(
        wx, wWihF_bf, wWihB_bf, wfbih, wfbhh, wbbih, wbbhh, xgF, xgB);

    // sequential word-level scan (latency-bound, batch=1)
    word_lstm<<<2, 1024, 0, stream>>>(xgF, xgB, wfWhh, wbWhh, out_hs);

    // projections + log_softmax
    proj_logsoftmax<<<NW, 128, 0, stream>>>(out_hs, Wp, bp, Wp2, bp2, (float*)d_out);
}
